// RNNDecoder_28647431864741
// MI455X (gfx1250) — matrix operand order repaired and checked by reading, not compiled, not run on hardware
//
#include <hip/hip_runtime.h>
#include <hip/hip_bf16.h>

// ---------------- problem constants ----------------
#define B_   2048
#define DIN  512
#define E_   256
#define H_   512
#define V_   1000
#define L_   64
#define TM   16     // batch rows per workgroup
#define NW   8      // wave32s per workgroup (256 threads)

typedef __attribute__((ext_vector_type(16))) __bf16          v16bf;
typedef __attribute__((ext_vector_type(8)))  float           v8f;
typedef __attribute__((ext_vector_type(8)))  unsigned short  v8u;

union FragU { v16bf f; v8u h[2]; };

__device__ __forceinline__ unsigned short f2bf(float x) {
    union { float f; unsigned u; } c; c.f = x;
    unsigned r = c.u + 0x7FFFu + ((c.u >> 16) & 1u);   // round-to-nearest-even
    return (unsigned short)(r >> 16);
}
__device__ __forceinline__ float bf2f(unsigned short h) {
    union { float f; unsigned u; } c; c.u = ((unsigned)h) << 16; return c.f;
}

// D = A*B + C  (bf16 inputs, f32 accumulate)
__device__ __forceinline__ v8f wmma_bf16(v16bf a, v16bf b, v8f c) {
    return __builtin_amdgcn_wmma_f32_16x16x32_bf16(
        /*neg_a=*/false, a, /*neg_b=*/false, b,
        /*c_mod=*/(short)0, c, /*reuse_a=*/false, /*reuse_b=*/false);
}

// A-matrix 16x32 bf16 fragment, documented CDNA5 layout:
//   lanes 0-15 : row m=lane,    K = {k0..k0+7, k0+16..k0+23}
//   lanes16-31 : row m=lane-16, K = {k0+8..k0+15, k0+24..k0+31}
__device__ __forceinline__ v16bf load_a(const unsigned short* base, int ld,
                                        int row0, int k0, int lane) {
    int m = lane & 15, hf = lane >> 4;
    const unsigned short* p = base + (size_t)(row0 + m) * ld + k0 + hf * 8;
    FragU u;
    u.h[0] = *(const v8u*)p;         // 8 halves at k0 + hf*8
    u.h[1] = *(const v8u*)(p + 16);  // 8 halves at k0 + 16 + hf*8
    return u.f;
}

// B-matrix 32x16 bf16 fragment; B[k][n] = W[n0+n][k] (W row-major [N][K]):
//   lanes 0-15 hold K={k0..k0+7, k0+16..k0+23} for N=lane ; lanes 16-31 hold K={k0+8..k0+15, k0+24..k0+31}
__device__ __forceinline__ v16bf load_b(const unsigned short* W, int ldk,
                                        int n0, int k0, int lane, int maxrow) {
    int n = lane & 15, hf = lane >> 4;
    int row = n0 + n; if (row >= maxrow) row = maxrow - 1;
    const unsigned short* p = W + (size_t)row * ldk + k0 + hf * 8;
    FragU u;
    u.h[0] = *(const v8u*)p;
    u.h[1] = *(const v8u*)(p + 16);
    return u.f;
}

// ---------------- fp32 -> bf16 conversion ----------------
__global__ void cvt_bf16_kernel(const float* __restrict__ src,
                                unsigned short* __restrict__ dst, int n) {
    int i = blockIdx.x * 256 + threadIdx.x;
    if (i < n) dst[i] = f2bf(src[i]);
}

// ---------------- persistent GRU decoder ----------------
__global__ void __launch_bounds__(256)
gru_decoder_kernel(const unsigned short* __restrict__ xbf,
                   const unsigned short* __restrict__ preWbf,
                   const unsigned short* __restrict__ Wihbf,
                   const unsigned short* __restrict__ Whhbf,
                   const unsigned short* __restrict__ proWbf,
                   const unsigned short* __restrict__ embbf,
                   const unsigned short* __restrict__ sosbf,
                   const float* __restrict__ pre_b,
                   const float* __restrict__ b_ih,
                   const float* __restrict__ b_hh,
                   const float* __restrict__ pro_b,
                   float* __restrict__ out_tok,      // [B,L] tokens as float
                   float* __restrict__ out_logits)   // [B,L,V]
{
    __shared__ unsigned short h_s[2][TM][H_];   // double-buffered hidden state (bf16)
    __shared__ unsigned short i_s[TM][E_];      // current input embedding (bf16)
    __shared__ float amax_v[NW * 32 * 8];
    __shared__ int   amax_i[NW * 32 * 8];
    __shared__ int   tok_s[TM];

    const int tid  = threadIdx.x;
    const int lane = tid & 31;
    const int wave = tid >> 5;
    const int b0   = blockIdx.x * TM;
    const int ncol = lane & 15;
    const int hf   = lane >> 4;

    // ---------- prologue: h0 = x @ pre_W^T + pre_b ----------
    for (int t = wave; t < H_ / 16; t += NW) {
        int n0 = t * 16;
        v8f acc = {};
        for (int k = 0; k < DIN; k += 32) {
            v16bf a = load_a(xbf, DIN, b0, k, lane);
            v16bf b = load_b(preWbf, DIN, n0, k, lane, H_);
            acc = wmma_bf16(a, b, acc);
        }
        float bias = pre_b[n0 + ncol];
        #pragma unroll
        for (int r = 0; r < 8; ++r) {
            int m = r + 8 * hf;
            h_s[0][m][n0 + ncol] = f2bf(acc[r] + bias);
        }
    }
    // i0 = sos_embed broadcast to all TM rows
    for (int idx = tid; idx < TM * E_; idx += 256)
        i_s[idx / E_][idx % E_] = sosbf[idx % E_];
    __syncthreads();

    int cur = 0;
    for (int l = 0; l < L_; ++l) {
        const int nxt = cur ^ 1;

        // ---------- phase 1: GRU gates + state update ----------
        for (int t = wave; t < H_ / 16; t += NW) {      // 32 column tiles
            int n0 = t * 16;
            v8f ar = {}, az = {}, ain = {}, ahn = {};
            // input contribution (K = E)
            for (int k = 0; k < E_; k += 32) {
                v16bf a = load_a(&i_s[0][0], E_, 0, k, lane);
                ar  = wmma_bf16(a, load_b(Wihbf, E_,          n0, k, lane, 3 * H_), ar);
                az  = wmma_bf16(a, load_b(Wihbf, E_,    H_ +  n0, k, lane, 3 * H_), az);
                ain = wmma_bf16(a, load_b(Wihbf, E_, 2 * H_ + n0, k, lane, 3 * H_), ain);
            }
            // hidden contribution (K = H)
            for (int k = 0; k < H_; k += 32) {
                v16bf a = load_a(&h_s[cur][0][0], H_, 0, k, lane);
                ar  = wmma_bf16(a, load_b(Whhbf, H_,          n0, k, lane, 3 * H_), ar);
                az  = wmma_bf16(a, load_b(Whhbf, H_,    H_ +  n0, k, lane, 3 * H_), az);
                ahn = wmma_bf16(a, load_b(Whhbf, H_, 2 * H_ + n0, k, lane, 3 * H_), ahn);
            }
            const int col = n0 + ncol;
            const float brz = b_ih[col] + b_hh[col];
            const float bzz = b_ih[H_ + col] + b_hh[H_ + col];
            const float bin = b_ih[2 * H_ + col];
            const float bhn = b_hh[2 * H_ + col];
            #pragma unroll
            for (int r = 0; r < 8; ++r) {
                int m = r + 8 * hf;
                float rg = 1.0f / (1.0f + expf(-(ar[r] + brz)));
                float zg = 1.0f / (1.0f + expf(-(az[r] + bzz)));
                float ng = tanhf(ain[r] + bin + rg * (ahn[r] + bhn));
                float ho = bf2f(h_s[cur][m][col]);
                float hn = (1.0f - zg) * ng + zg * ho;
                h_s[nxt][m][col] = f2bf(hn);
            }
        }
        __syncthreads();

        // ---------- phase 2: logits = h_new @ pro_W^T + pro_b, running argmax ----------
        float bestv[8]; int besti[8];
        #pragma unroll
        for (int r = 0; r < 8; ++r) { bestv[r] = -3.4e38f; besti[r] = 0; }

        for (int t = wave; t < (V_ + 15) / 16; t += NW) {   // 63 tiles
            int n0 = t * 16;
            v8f acc = {};
            for (int k = 0; k < H_; k += 32) {
                v16bf a = load_a(&h_s[nxt][0][0], H_, 0, k, lane);
                acc = wmma_bf16(a, load_b(proWbf, H_, n0, k, lane, V_), acc);
            }
            const int col = n0 + ncol;
            const bool ok = (col < V_);
            const float pb = ok ? pro_b[col] : 0.0f;
            #pragma unroll
            for (int r = 0; r < 8; ++r) {
                int m = r + 8 * hf;
                float v = acc[r] + pb;
                if (ok) {
                    out_logits[((size_t)(b0 + m) * L_ + l) * V_ + col] = v;
                    if (v > bestv[r] || (v == bestv[r] && col < besti[r])) {
                        bestv[r] = v; besti[r] = col;
                    }
                }
            }
        }
        #pragma unroll
        for (int r = 0; r < 8; ++r) {
            amax_v[(wave * 32 + lane) * 8 + r] = bestv[r];
            amax_i[(wave * 32 + lane) * 8 + r] = besti[r];
        }
        __syncthreads();

        // ---------- phase 3: per-row argmax reduce + token emit ----------
        if (tid < TM) {
            int m = tid;
            int r = m & 7, mh = m >> 3;
            float bv = -3.4e38f; int bi = 0;
            for (int w = 0; w < NW; ++w)
                for (int ln = mh * 16; ln < mh * 16 + 16; ++ln) {
                    float v = amax_v[(w * 32 + ln) * 8 + r];
                    int   i2 = amax_i[(w * 32 + ln) * 8 + r];
                    if (v > bv || (v == bv && i2 < bi)) { bv = v; bi = i2; }
                }
            tok_s[m] = bi;
            out_tok[(size_t)(b0 + m) * L_ + l] = (float)bi;
        }
        __syncthreads();

        // ---------- embedding gather: i_next = embed_table[tok] ----------
        {
            int row = tid >> 4, chunk = tid & 15;       // 16 rows x 16 chunks of 16 halves
            const unsigned short* src = embbf + (size_t)tok_s[row] * E_ + chunk * 16;
            v8u a0 = *(const v8u*)src;
            v8u a1 = *(const v8u*)(src + 8);
            *(v8u*)&i_s[row][chunk * 16]     = a0;
            *(v8u*)&i_s[row][chunk * 16 + 8] = a1;
        }
        cur ^= 1;
        __syncthreads();
    }
}

// ---------------- host launcher ----------------
extern "C" void kernel_launch(void* const* d_in, const int* in_sizes, int n_in,
                              void* d_out, int out_size, void* d_ws, size_t ws_size,
                              hipStream_t stream) {
    (void)in_sizes; (void)n_in; (void)out_size; (void)ws_size;

    const float* x    = (const float*)d_in[0];
    const float* sos  = (const float*)d_in[1];
    const float* emb  = (const float*)d_in[2];
    const float* preW = (const float*)d_in[3];
    const float* preb = (const float*)d_in[4];
    const float* Wih  = (const float*)d_in[5];
    const float* Whh  = (const float*)d_in[6];
    const float* bih  = (const float*)d_in[7];
    const float* bhh  = (const float*)d_in[8];
    const float* proW = (const float*)d_in[9];
    const float* prob = (const float*)d_in[10];

    // bf16 workspace layout (~6.5 MB)
    unsigned short* ws = (unsigned short*)d_ws;
    size_t off = 0;
    unsigned short* xbf    = ws + off; off += (size_t)B_ * DIN;      // x
    unsigned short* preWbf = ws + off; off += (size_t)H_ * DIN;      // pre_W
    unsigned short* Wihbf  = ws + off; off += (size_t)3 * H_ * E_;   // W_ih
    unsigned short* Whhbf  = ws + off; off += (size_t)3 * H_ * H_;   // W_hh
    unsigned short* proWbf = ws + off; off += (size_t)V_ * H_;       // pro_W
    unsigned short* embbf  = ws + off; off += (size_t)V_ * E_;       // embed_table
    unsigned short* sosbf  = ws + off; off += (size_t)E_;            // sos_embed

    struct CvtJob { const float* s; unsigned short* d; int n; };
    const CvtJob jobs[7] = {
        { x,    xbf,    B_ * DIN },
        { preW, preWbf, H_ * DIN },
        { Wih,  Wihbf,  3 * H_ * E_ },
        { Whh,  Whhbf,  3 * H_ * H_ },
        { proW, proWbf, V_ * H_ },
        { emb,  embbf,  V_ * E_ },
        { sos,  sosbf,  E_ },
    };
    for (int j = 0; j < 7; ++j)
        cvt_bf16_kernel<<<(jobs[j].n + 255) / 256, 256, 0, stream>>>(
            jobs[j].s, jobs[j].d, jobs[j].n);

    float* out_tok    = (float*)d_out;                 // [B,L] tokens (as float)
    float* out_logits = out_tok + (size_t)B_ * L_;     // [B,L,V]

    gru_decoder_kernel<<<B_ / TM, NW * 32, 0, stream>>>(
        xbf, preWbf, Wihbf, Whhbf, proWbf, embbf, sosbf,
        preb, bih, bhh, prob, out_tok, out_logits);
}
